// Generator_54769422959369
// MI455X (gfx1250) — matrix operand order repaired and checked by reading, not compiled, not run on hardware
//
#include <hip/hip_runtime.h>

typedef __bf16 bf16;
typedef __attribute__((ext_vector_type(8)))  bf16  v8bf;
typedef __attribute__((ext_vector_type(16))) bf16  v16bf;
typedef __attribute__((ext_vector_type(8)))  float v8f;

#define CAT16(lo, hi) __builtin_shufflevector(lo, hi, 0,1,2,3,4,5,6,7,8,9,10,11,12,13,14,15)

static constexpr int NROW = 4096;   // batch
static constexpr int CCOL = 100;    // columns
static constexpr int DEMB = 8;      // datum emb
static constexpr int KIN  = CCOL * DEMB;   // 800
static constexpr int FH   = 512;    // hidden
static constexpr int HCOL = 16;     // per-column hidden
static constexpr int KMID = CCOL * HCOL;   // 1600
static constexpr int OUT  = 2;

__device__ __forceinline__ bf16 f2bf(float f) {
    unsigned u = __builtin_bit_cast(unsigned, f);
    unsigned r = u + 0x7FFFu + ((u >> 16) & 1u);   // round-to-nearest-even
    unsigned short h = (unsigned short)(r >> 16);
    return __builtin_bit_cast(bf16, h);
}

__device__ __forceinline__ v8f bf16_wmma(v16bf a, v16bf b, v8f c) {
    // D = A(16x32) * B(32x16) + C, f32 accumulate
    return __builtin_amdgcn_wmma_f32_16x16x32_bf16(false, a, false, b,
                                                   (short)0, c, false, false);
}

// async global -> LDS copy of 16 bytes per active lane (ASYNCcnt-tracked)
__device__ __forceinline__ void async_copy_b128(unsigned lds_byte_addr,
                                                const void* gaddr) {
    asm volatile("global_load_async_to_lds_b128 %0, %1, off"
                 :: "v"(lds_byte_addr), "v"(gaddr) : "memory");
}
__device__ __forceinline__ void wait_async0() {
    asm volatile("s_wait_asynccnt 0x0" ::: "memory");
}

// ---------------------------------------------------------------------------
// Prep: xinfo[N,800] = x[n,c]*w_data[c,:] + emb[mask*(c+1), :]   (bf16)
// ---------------------------------------------------------------------------
__global__ void k_prep_xinfo(const float* __restrict__ x, const int* __restrict__ mask,
                             const float* __restrict__ w_data, const float* __restrict__ emb,
                             bf16* __restrict__ xinfo) {
    int id = blockIdx.x * blockDim.x + threadIdx.x;
    if (id >= NROW * CCOL) return;
    int n = id / CCOL, c = id % CCOL;
    float xv = x[id];
    int idx = mask[id] * (c + 1);          // padding row 0 is zeros
    v8bf v;
#pragma unroll
    for (int d = 0; d < DEMB; ++d)
        v[d] = f2bf(xv * w_data[c * DEMB + d] + emb[idx * DEMB + d]);
    *(v8bf*)(xinfo + (size_t)n * KIN + c * DEMB) = v;
}

// ---------------------------------------------------------------------------
// Weight transposes -> bf16 [Nfeat][K] so B-operand loads are contiguous
// ---------------------------------------------------------------------------
__global__ void k_t_xw1(const float* __restrict__ w, bf16* __restrict__ wt) {
    int id = blockIdx.x * blockDim.x + threadIdx.x;
    if (id >= FH * KIN) return;
    int nf = id / KIN, k = id % KIN;
    wt[id] = f2bf(w[(size_t)k * FH + nf]);          // [512][800]
}
__global__ void k_t_xw2(const float* __restrict__ w, bf16* __restrict__ wt) {
    int id = blockIdx.x * blockDim.x + threadIdx.x;
    if (id >= KMID * FH) return;
    int nf = id / FH, k = id % FH;
    wt[id] = f2bf(w[(size_t)k * KMID + nf]);        // [1600][512]
}
__global__ void k_t_cw1(const float* __restrict__ w, bf16* __restrict__ wt) {
    int id = blockIdx.x * blockDim.x + threadIdx.x;
    if (id >= CCOL * FH * HCOL) return;
    int c = id / (FH * HCOL), r = id % (FH * HCOL);
    int f = r / HCOL, h = r % HCOL;
    wt[id] = f2bf(w[(size_t)c * HCOL * FH + h * FH + f]);   // [100][512][16]
}
__global__ void k_t_cw2(const float* __restrict__ w, bf16* __restrict__ wt) {
    int id = blockIdx.x * blockDim.x + threadIdx.x;
    if (id >= CCOL * 16 * FH) return;
    int c = id / (16 * FH), r = id % (16 * FH);
    int n = r / FH, k = r % FH;                      // pad O=2 -> 16 cols
    wt[id] = (n < OUT) ? f2bf(w[(size_t)c * FH * OUT + k * OUT + n]) : f2bf(0.0f);
}

// ---------------------------------------------------------------------------
// GEMM1 [N,800]x[800,512] + bias -> LN -> ReLU -> bf16 [N,512]
// block: 16 rows, 256 thr (8 waves x 4 ntiles = 512 feats)
// A-tile (16x800 bf16 = 25.6 KB, contiguous) staged LDS via async copy once,
// then every wave reads A with ds_load_b128 instead of 8x-redundant VMEM.
// ---------------------------------------------------------------------------
__global__ void __launch_bounds__(256) k_gemm1_ln(
        const bf16* __restrict__ A, const bf16* __restrict__ Bt,
        const float* __restrict__ bias, const float* __restrict__ g,
        const float* __restrict__ be, bf16* __restrict__ outv) {
    __shared__ bf16  Atile[16 * KIN];          // 25600 B
    __shared__ float fbuf[16 * FH];            // 32768 B
    __shared__ float psum[16][16], psq[16][16];
    __shared__ float mu_s[16], rs_s[16];
    const int tid = threadIdx.x;
    const int wave = tid >> 5, lane = tid & 31;
    const int hi = lane >> 4, lr = lane & 15;
    const int row0 = blockIdx.x * 16;
    const int nbase = wave * 64;

    // ---- async stage of the whole A block (16 contiguous rows) ----
    {
        const char* gbase = (const char*)(A + (size_t)row0 * KIN);
        unsigned ldsbase = (unsigned)(unsigned long long)(void*)Atile;
        for (int o = tid * 16; o < 16 * KIN * 2; o += 256 * 16)
            async_copy_b128(ldsbase + (unsigned)o, gbase + o);
        wait_async0();
    }
    __syncthreads();

    v8f acc[4] = {};
    const bf16* Arow = Atile + lr * KIN;       // LDS reads
    for (int k0 = 0; k0 < KIN; k0 += 32) {
        v8bf alo = *(const v8bf*)(Arow + k0 + 8 * hi);
        v8bf ah  = *(const v8bf*)(Arow + k0 + 16 + 8 * hi);
        v16bf a = CAT16(alo, ah);
#pragma unroll
        for (int nt = 0; nt < 4; ++nt) {
            int col = nbase + nt * 16 + lr;
            const bf16* Brow = Bt + (size_t)col * KIN + k0 + 8 * hi;
            v8bf b0 = *(const v8bf*)Brow;
            v8bf b1 = *(const v8bf*)(Brow + 16);
            acc[nt] = bf16_wmma(a, CAT16(b0, b1), acc[nt]);
        }
    }
#pragma unroll
    for (int nt = 0; nt < 4; ++nt) {
        int col = nbase + nt * 16 + lr;
        float bia = bias[col];
#pragma unroll
        for (int i = 0; i < 8; ++i)
            fbuf[(i + 8 * hi) * FH + col] = acc[nt][i] + bia;
    }
    __syncthreads();

    int r = tid >> 4, sub = tid & 15;
    float s = 0.f, s2 = 0.f;
    const float* rp = fbuf + r * FH + sub * 32;
#pragma unroll
    for (int j = 0; j < 32; ++j) { float v = rp[j]; s += v; s2 += v * v; }
    psum[r][sub] = s; psq[r][sub] = s2;
    __syncthreads();
    if (tid < 16) {
        float ss = 0.f, ss2 = 0.f;
#pragma unroll
        for (int j = 0; j < 16; ++j) { ss += psum[tid][j]; ss2 += psq[tid][j]; }
        float mu = ss * (1.f / FH);
        float var = ss2 * (1.f / FH) - mu * mu;
        mu_s[tid] = mu; rs_s[tid] = rsqrtf(var + 1e-5f);
    }
    __syncthreads();
    float mu = mu_s[r], rs = rs_s[r];
    bf16* orow = outv + (size_t)(row0 + r) * FH;
#pragma unroll
    for (int j = 0; j < 32; ++j) {
        int f = sub * 32 + j;
        float v = (fbuf[r * FH + f] - mu) * rs * g[f] + be[f];
        orow[f] = f2bf(fmaxf(v, 0.f));
    }
}

// ---------------------------------------------------------------------------
// GEMM2 [N,512]x[512,1600] + bias -> bf16 [N,1600]
// grid (N/16, 25), 128 thr = 4 waves, one 16x16 tile each
// ---------------------------------------------------------------------------
__global__ void __launch_bounds__(128) k_gemm2(
        const bf16* __restrict__ A, const bf16* __restrict__ Bt,
        const float* __restrict__ bias, bf16* __restrict__ outv) {
    const int tid = threadIdx.x;
    const int wave = tid >> 5, lane = tid & 31;
    const int hi = lane >> 4, lr = lane & 15;
    const int row0 = blockIdx.x * 16;
    const int n0 = (blockIdx.y * 4 + wave) * 16;

    v8f acc = {};
    const bf16* Arow = A + (size_t)(row0 + lr) * FH;
    const bf16* Brow = Bt + (size_t)(n0 + lr) * FH;
    for (int k0 = 0; k0 < FH; k0 += 32) {
        v8bf alo = *(const v8bf*)(Arow + k0 + 8 * hi);
        v8bf ah  = *(const v8bf*)(Arow + k0 + 16 + 8 * hi);
        v8bf b0  = *(const v8bf*)(Brow + k0 + 8 * hi);
        v8bf b1  = *(const v8bf*)(Brow + k0 + 16 + 8 * hi);
        acc = bf16_wmma(CAT16(alo, ah), CAT16(b0, b1), acc);
    }
    int col = n0 + lr;
    float bia = bias[col];
    bf16* op = outv + (size_t)row0 * KMID + col;
#pragma unroll
    for (int i = 0; i < 8; ++i)
        op[(size_t)(i + 8 * hi) * KMID] = f2bf(acc[i] + bia);
}

// ---------------------------------------------------------------------------
// Per-column FFN, fused: [16,16]x[16,512]+b -> LN -> ReLU -> x[512,2]+b
// grid (N/16, 100), 256 thr. h2 never leaves LDS.
// ---------------------------------------------------------------------------
__global__ void __launch_bounds__(256) k_col_ffn(
        const bf16* __restrict__ Xo, const bf16* __restrict__ W1t,
        const float* __restrict__ b1, const float* __restrict__ g,
        const float* __restrict__ be, const bf16* __restrict__ W2t,
        const float* __restrict__ b2, float* __restrict__ outp) {
    __shared__ float fbuf[16 * FH];
    __shared__ bf16  h2buf[16 * FH];
    __shared__ float red[8 * 256];
    __shared__ float psum[16][16], psq[16][16];
    __shared__ float mu_s[16], rs_s[16];
    const int c = blockIdx.y;
    const int row0 = blockIdx.x * 16;
    const int tid = threadIdx.x;
    const int wave = tid >> 5, lane = tid & 31;
    const int hi = lane >> 4, lr = lane & 15;
    const int nbase = wave * 64;

    // --- GEMM h2 = Xo[:, c*16:+16] (K=16, zero-padded to 32) x cw1t[c] ---
    v8bf z8 = {};
    v8bf alo = *(const v8bf*)(Xo + (size_t)(row0 + lr) * KMID + c * HCOL + 8 * hi);
    v16bf a = CAT16(alo, z8);            // K=16..31 are zero
    v8f acc[4] = {};
#pragma unroll
    for (int nt = 0; nt < 4; ++nt) {
        int f = nbase + nt * 16 + lr;
        v16bf b = {};
        {                                // half h holds K=8h..8h+7; K=16..31 -> zero
            const bf16* Bp = W1t + (size_t)c * FH * HCOL + f * HCOL;
            v8bf b0 = *(const v8bf*)(Bp + 8 * hi);
            b = CAT16(b0, z8);
        }
        acc[nt] = bf16_wmma(a, b, acc[nt]);
    }
#pragma unroll
    for (int nt = 0; nt < 4; ++nt) {
        int f = nbase + nt * 16 + lr;
        float bia = b1[c * FH + f];
#pragma unroll
        for (int i = 0; i < 8; ++i)
            fbuf[(i + 8 * hi) * FH + f] = acc[nt][i] + bia;
    }
    __syncthreads();

    // --- LN (per-column gamma/beta) + ReLU -> bf16 h2buf ---
    int r = tid >> 4, sub = tid & 15;
    float s = 0.f, s2 = 0.f;
    const float* rp = fbuf + r * FH + sub * 32;
#pragma unroll
    for (int j = 0; j < 32; ++j) { float v = rp[j]; s += v; s2 += v * v; }
    psum[r][sub] = s; psq[r][sub] = s2;
    __syncthreads();
    if (tid < 16) {
        float ss = 0.f, ss2 = 0.f;
#pragma unroll
        for (int j = 0; j < 16; ++j) { ss += psum[tid][j]; ss2 += psq[tid][j]; }
        float mu = ss * (1.f / FH);
        float var = ss2 * (1.f / FH) - mu * mu;
        mu_s[tid] = mu; rs_s[tid] = rsqrtf(var + 1e-5f);
    }
    __syncthreads();
    {
        float mu = mu_s[r], rs = rs_s[r];
#pragma unroll
        for (int j = 0; j < 32; ++j) {
            int f = sub * 32 + j;
            float v = (fbuf[r * FH + f] - mu) * rs * g[c * FH + f] + be[c * FH + f];
            h2buf[r * FH + f] = f2bf(fmaxf(v, 0.f));
        }
    }
    __syncthreads();

    // --- head GEMM: h2[16,512] x cw2t[c] (padded [16][512]); K split 8 waves ---
    v8f acc2 = {};
    const bf16* hrow = h2buf + lr * FH;                    // A row (LDS)
    const bf16* Brow = W2t + (size_t)c * 16 * FH + lr * FH; // B col (padded)
    const int kbase = wave * 64;
#pragma unroll
    for (int kk = 0; kk < 64; kk += 32) {
        int k0 = kbase + kk;
        v8bf a0 = *(const v8bf*)(hrow + k0 + 8 * hi);
        v8bf a1 = *(const v8bf*)(hrow + k0 + 16 + 8 * hi);
        v8bf b0 = *(const v8bf*)(Brow + k0 + 8 * hi);
        v8bf bb = *(const v8bf*)(Brow + k0 + 16 + 8 * hi);
        acc2 = bf16_wmma(CAT16(a0, a1), CAT16(b0, bb), acc2);
    }
#pragma unroll
    for (int i = 0; i < 8; ++i)
        red[wave * 256 + (i + 8 * hi) * 16 + lr] = acc2[i];
    __syncthreads();

    int m = tid >> 4, n = tid & 15;
    if (n < OUT) {
        float sum = 0.f;
#pragma unroll
        for (int w = 0; w < 8; ++w) sum += red[w * 256 + m * 16 + n];
        outp[(size_t)(row0 + m) * (CCOL * OUT) + c * OUT + n] = sum + b2[c * OUT + n];
    }
}

// ---------------------------------------------------------------------------
extern "C" void kernel_launch(void* const* d_in, const int* in_sizes, int n_in,
                              void* d_out, int out_size, void* d_ws, size_t ws_size,
                              hipStream_t stream) {
    (void)in_sizes; (void)n_in; (void)out_size; (void)ws_size;
    const float* x      = (const float*)d_in[0];
    const int*   mask   = (const int*)  d_in[1];
    const float* w_data = (const float*)d_in[2];
    const float* emb    = (const float*)d_in[3];
    const float* xw1    = (const float*)d_in[4];
    const float* xb1    = (const float*)d_in[5];
    const float* xg     = (const float*)d_in[6];
    const float* xbe    = (const float*)d_in[7];
    const float* xw2    = (const float*)d_in[8];
    const float* xb2    = (const float*)d_in[9];
    const float* cw1    = (const float*)d_in[10];
    const float* cb1    = (const float*)d_in[11];
    const float* cg     = (const float*)d_in[12];
    const float* cbe    = (const float*)d_in[13];
    const float* cw2    = (const float*)d_in[14];
    const float* cb2    = (const float*)d_in[15];
    float* out = (float*)d_out;

    char* w = (char*)d_ws;
    bf16* xinfo  = (bf16*)w; w += (size_t)NROW * KIN * 2;        // 6.55 MB
    bf16* xw1t   = (bf16*)w; w += (size_t)FH * KIN * 2;          // 0.82 MB
    bf16* hrelu  = (bf16*)w; w += (size_t)NROW * FH * 2;         // 4.19 MB
    bf16* xw2t   = (bf16*)w; w += (size_t)KMID * FH * 2;         // 1.64 MB
    bf16* xout   = (bf16*)w; w += (size_t)NROW * KMID * 2;       // 13.1 MB
    bf16* cw1t   = (bf16*)w; w += (size_t)CCOL * FH * HCOL * 2;  // 1.64 MB
    bf16* cw2t   = (bf16*)w; w += (size_t)CCOL * 16 * FH * 2;    // 1.64 MB

    k_prep_xinfo<<<(NROW * CCOL + 255) / 256, 256, 0, stream>>>(x, mask, w_data, emb, xinfo);
    k_t_xw1<<<(FH * KIN + 255) / 256, 256, 0, stream>>>(xw1, xw1t);
    k_t_xw2<<<(KMID * FH + 255) / 256, 256, 0, stream>>>(xw2, xw2t);
    k_t_cw1<<<(CCOL * FH * HCOL + 255) / 256, 256, 0, stream>>>(cw1, cw1t);
    k_t_cw2<<<(CCOL * 16 * FH + 255) / 256, 256, 0, stream>>>(cw2, cw2t);

    k_gemm1_ln<<<NROW / 16, 256, 0, stream>>>(xinfo, xw1t, xb1, xg, xbe, hrelu);
    k_gemm2<<<dim3(NROW / 16, KMID / 64), 128, 0, stream>>>(hrelu, xw2t, xb2, xout);
    k_col_ffn<<<dim3(NROW / 16, CCOL), 256, 0, stream>>>(xout, cw1t, cb1, cg, cbe,
                                                         cw2t, cb2, out);
}
